// CliffordLISTA_33243046871243
// MI455X (gfx1250) — matrix operand order repaired and checked by reading, not compiled, not run on hardware
//
#include <hip/hip_runtime.h>
#include <hip/hip_bf16.h>

// ---------------------------------------------------------------------------
// CliffordLISTA on MI455X (gfx1250):
//   geometric_product == one big GEMM with sign-permuted weights (Cayley table
//   has exactly one nonzero per (i,j)).  All GEMMs run through
//   v_wmma_f32_16x16x32_bf16 (bf16 operands, f32 accumulate), with
//   double-buffered GLOBAL_LOAD_ASYNC_TO_LDS_B128 DMA staging.
//   Wave tile 64x64 (16 WMMA / K-step from 8 fragment loads).
// Problem sizes: B=2048, IN=256, HID=512, NB=8 -> K1=2048, K2=Nc=4096.
// ---------------------------------------------------------------------------

typedef __attribute__((ext_vector_type(16))) __bf16 v16bf;
typedef __attribute__((ext_vector_type(8)))  __bf16 v8bf;
typedef __attribute__((ext_vector_type(8)))  float  v8f;
typedef __attribute__((ext_vector_type(4)))  int    v4i;

#define TM 128
#define TN 256
#define TK 32
#define LDSS 40   // padded LDS row stride in halves (80B -> conflict-free b128)

#if __has_builtin(__builtin_amdgcn_global_load_async_to_lds_b128)
#define HAVE_ASYNC_LDS 1
#else
#define HAVE_ASYNC_LDS 0
#endif

#if HAVE_ASYNC_LDS
typedef __attribute__((address_space(1))) v4i gas_v4i;  // global (AS1) int4
typedef __attribute__((address_space(3))) v4i las_v4i;  // LDS (AS3) int4

__device__ __forceinline__ void async_ld_b128(const __bf16* gsrc, __bf16* ldst) {
    __builtin_amdgcn_global_load_async_to_lds_b128(
        (gas_v4i*)gsrc, (las_v4i*)ldst, /*offset=*/0, /*cpol=*/0);
}
template <int N>
__device__ __forceinline__ void wait_asynccnt() {
#if __has_builtin(__builtin_amdgcn_s_wait_asynccnt)
    __builtin_amdgcn_s_wait_asynccnt(N);
#else
    asm volatile("s_wait_asynccnt %0" ::"i"(N) : "memory");
#endif
}
#endif

// Reordering sign of e_i * e_j for Euclidean signature (SIG_G = all +1).
__device__ __forceinline__ float cayley_sign(int a, int b) {
    int cnt = 0;
    for (int aa = a >> 1; aa; aa >>= 1) cnt += __builtin_popcount(aa & b);
    return (cnt & 1) ? -1.0f : 1.0f;
}

// BT[(h*8+k)*K + (n*8+i)] = s(i, i^k) * W[h, n, i^k]   (bf16, K-contiguous)
__global__ __launch_bounds__(256)
void prep_weights_kernel(const float* __restrict__ W, __bf16* __restrict__ BT,
                         int Nin, int K, int total) {
    int gid = blockIdx.x * blockDim.x + threadIdx.x;
    if (gid >= total) return;
    int kk = gid & (K - 1);          // K is a power of two (2048 / 4096)
    int nc = gid / K;
    int h = nc >> 3, k = nc & 7;
    int n = kk >> 3, i = kk & 7;
    int j = i ^ k;
    float s = cayley_sign(i, j);
    float v = s * W[(((size_t)h * Nin + n) << 3) + j];
    BT[gid] = (__bf16)v;
}

__global__ __launch_bounds__(256)
void f32_to_bf16_kernel(const float* __restrict__ src, __bf16* __restrict__ dst,
                        int total) {
    int gid = blockIdx.x * blockDim.x + threadIdx.x;
    if (gid >= total) return;
    dst[gid] = (__bf16)src[gid];
}

// Layer 0: x = soft_threshold(W1y)   (x_prev == 0)
__global__ __launch_bounds__(256)
void lista_first_kernel(const float* __restrict__ W1y,
                        const float* __restrict__ lambdas,
                        float* __restrict__ xf, __bf16* __restrict__ xbf,
                        int total) {
    int gid = blockIdx.x * blockDim.x + threadIdx.x;
    if (gid >= total) return;
    float z = W1y[gid];
    float lam = lambdas[__builtin_popcount(gid & 7)];
    float a = fabsf(z) - lam;
    a = a > 0.0f ? a : 0.0f;
    float x = copysignf(a, z);
    xf[gid]  = x;
    xbf[gid] = (__bf16)x;
}

// Per-wave compute of one K-step: 4 A + 4 B fragments from LDS, 16 WMMAs.
__device__ __forceinline__ void mma_step(const __bf16* __restrict__ Asb,
                                         const __bf16* __restrict__ Bsb,
                                         int wm0, int wn0, int lane16,
                                         int aklo, int bklo, v8f (&acc)[4][4]) {
    v16bf afrag[4], bfrag[4];
#pragma unroll
    for (int t = 0; t < 4; ++t) {
        // A 16x32 bf16 layout: lanes0-15 K{0..7,16..23}, lanes16-31 K{8..15,24..31}
        const __bf16* p = &Asb[(wm0 + t * 16 + lane16) * LDSS];
        v8bf lo = *(const v8bf*)(p + aklo);
        v8bf hi = *(const v8bf*)(p + aklo + 16);
        afrag[t] = __builtin_shufflevector(lo, hi, 0, 1, 2, 3, 4, 5, 6, 7,
                                           8, 9, 10, 11, 12, 13, 14, 15);
    }
#pragma unroll
    for (int u = 0; u < 4; ++u) {
        // B 32x16 bf16 layout: lanes0-15 K{0..7,16..23}, lanes16-31 K{8..15,24..31}
        const __bf16* p = &Bsb[(wn0 + u * 16 + lane16) * LDSS + bklo];
        v8bf lo = *(const v8bf*)p;
        v8bf hi = *(const v8bf*)(p + 16);
        bfrag[u] = __builtin_shufflevector(lo, hi, 0, 1, 2, 3, 4, 5, 6, 7,
                                           8, 9, 10, 11, 12, 13, 14, 15);
    }
#pragma unroll
    for (int t = 0; t < 4; ++t)
#pragma unroll
        for (int u = 0; u < 4; ++u)
            acc[t][u] = __builtin_amdgcn_wmma_f32_16x16x32_bf16(
                false, afrag[t], false, bfrag[u], (short)0, acc[t][u],
                false, false);
}

// C[M,N] = A[M,K] * BT[N,K]^T  with optional fused LISTA epilogue.
// mode 0: Cf = A*B                        (W1y pass)
// mode 1: x = softthr(bias + A*B); Cf = x (f32), Cbf = x (bf16)
__global__ __launch_bounds__(256)
void gemm_wmma_bf16(const __bf16* __restrict__ A,    // [M x K] row-major
                    const __bf16* __restrict__ BT,   // [N x K] row-major
                    const float*  __restrict__ bias, // [M x N] or null
                    float*        __restrict__ Cf,   // [M x N]
                    __bf16*       __restrict__ Cbf,  // [M x N] or null
                    const float*  __restrict__ lambdas,
                    int M, int N, int K, int mode) {
    const int tid    = threadIdx.x;
    const int lane   = tid & 31;
    const int wid    = tid >> 5;
    const int wm0    = (wid & 1) * 64;   // wave M origin (4 x 16)
    const int wn0    = (wid >> 1) * 64;  // wave N origin (4 x 16)
    const int blockM = blockIdx.y * TM;
    const int blockN = blockIdx.x * TN;

    const int lane16 = lane & 15;
    const int lhi    = lane >> 4;          // 0 or 1
    const int aklo   = 8 * lhi;            // A chunks: aklo, aklo+16
    const int bklo   = 8 * lhi;            // B chunks: bklo, bklo+16

    // staging: 4 threads per row, 8 halves (16B) each; A: 2 passes, B: 4 passes
    const int lrow = tid >> 2;             // 0..63
    const int lcol = (tid & 3) * 8;        // 0,8,16,24

    const __bf16* gA0 = A  + (size_t)(blockM + lrow) * K + lcol;
    const __bf16* gA1 = gA0 + (size_t)64 * K;
    const __bf16* gB0 = BT + (size_t)(blockN + lrow) * K + lcol;
    const __bf16* gB1 = gB0 + (size_t)64 * K;
    const __bf16* gB2 = gB0 + (size_t)128 * K;
    const __bf16* gB3 = gB0 + (size_t)192 * K;

    v8f acc[4][4] = {{}, {}, {}, {}};

#if HAVE_ASYNC_LDS
    // ---- double-buffered DMA staging (GLOBAL_LOAD_ASYNC_TO_LDS_B128) ----
    __shared__ __bf16 As[2][TM * LDSS];    // 2 x 10 KB
    __shared__ __bf16 Bs[2][TN * LDSS];    // 2 x 20 KB

    const int s0 = lrow * LDSS + lcol;
    const int s1 = s0 + 64 * LDSS;
    const int s2 = s0 + 128 * LDSS;
    const int s3 = s0 + 192 * LDSS;

    // prologue: kick off tile 0 into buffer 0
    async_ld_b128(gA0, &As[0][s0]);
    async_ld_b128(gA1, &As[0][s1]);
    async_ld_b128(gB0, &Bs[0][s0]);
    async_ld_b128(gB1, &Bs[0][s1]);
    async_ld_b128(gB2, &Bs[0][s2]);
    async_ld_b128(gB3, &Bs[0][s3]);

    const int nk = K / TK;
    for (int kt = 0; kt < nk; ++kt) {
        const int cur = kt & 1;
        if (kt + 1 < nk) {
            const int k1 = (kt + 1) * TK;   // DMA next tile while we compute
            async_ld_b128(gA0 + k1, &As[cur ^ 1][s0]);
            async_ld_b128(gA1 + k1, &As[cur ^ 1][s1]);
            async_ld_b128(gB0 + k1, &Bs[cur ^ 1][s0]);
            async_ld_b128(gB1 + k1, &Bs[cur ^ 1][s1]);
            async_ld_b128(gB2 + k1, &Bs[cur ^ 1][s2]);
            async_ld_b128(gB3 + k1, &Bs[cur ^ 1][s3]);
            wait_asynccnt<6>();             // tile kt's 6 DMAs done (in-order)
        } else {
            wait_asynccnt<0>();
        }
        __syncthreads();                    // tile kt visible to all waves
        mma_step(&As[cur][0], &Bs[cur][0], wm0, wn0, lane16, aklo, bklo, acc);
        __syncthreads();                    // reads done before buf reuse
    }
#else
    // ---- fallback: batched register staging, single buffer ----
    __shared__ __bf16 As[TM * LDSS];
    __shared__ __bf16 Bs[TN * LDSS];

    for (int k0 = 0; k0 < K; k0 += TK) {
        v8bf ra0 = *(const v8bf*)(gA0 + k0);
        v8bf ra1 = *(const v8bf*)(gA1 + k0);
        v8bf rb0 = *(const v8bf*)(gB0 + k0);
        v8bf rb1 = *(const v8bf*)(gB1 + k0);
        v8bf rb2 = *(const v8bf*)(gB2 + k0);
        v8bf rb3 = *(const v8bf*)(gB3 + k0);
        __syncthreads();                    // previous tile's reads complete
        *(v8bf*)&As[lrow * LDSS + lcol]         = ra0;
        *(v8bf*)&As[(lrow + 64) * LDSS + lcol]  = ra1;
        *(v8bf*)&Bs[lrow * LDSS + lcol]         = rb0;
        *(v8bf*)&Bs[(lrow + 64) * LDSS + lcol]  = rb1;
        *(v8bf*)&Bs[(lrow + 128) * LDSS + lcol] = rb2;
        *(v8bf*)&Bs[(lrow + 192) * LDSS + lcol] = rb3;
        __syncthreads();
        mma_step(&As[0], &Bs[0], wm0, wn0, lane16, aklo, bklo, acc);
    }
#endif

    // C layout: vgpr r, lane l -> m = 8*(l/16)+r, n = l%16
    const int mbase = blockM + wm0 + 8 * lhi;
#pragma unroll
    for (int t = 0; t < 4; ++t) {
#pragma unroll
        for (int u = 0; u < 4; ++u) {
            const int ncol = blockN + wn0 + u * 16 + lane16;
#pragma unroll
            for (int r = 0; r < 8; ++r) {
                const int m = mbase + 16 * t + r;
                const size_t idx = (size_t)m * N + ncol;
                float z = acc[t][u][r];
                if (mode == 0) {
                    Cf[idx] = z;
                } else {
                    z += bias[idx];
                    float lam = lambdas[__builtin_popcount(ncol & 7)];
                    float a = fabsf(z) - lam;
                    a = a > 0.0f ? a : 0.0f;
                    float x = copysignf(a, z);
                    Cf[idx]  = x;
                    Cbf[idx] = (__bf16)x;
                }
            }
        }
    }
}

// ---------------------------------------------------------------------------
extern "C" void kernel_launch(void* const* d_in, const int* in_sizes, int n_in,
                              void* d_out, int out_size, void* d_ws, size_t ws_size,
                              hipStream_t stream) {
    constexpr int Bm = 2048, IN = 256, HID = 512, NB = 8;
    constexpr int K1 = IN * NB;    // 2048
    constexpr int K2 = HID * NB;   // 4096
    constexpr int Nc = HID * NB;   // 4096
    constexpr int NLAYERS = 5;

    const float* y   = (const float*)d_in[0];
    const float* W1  = (const float*)d_in[1];
    const float* W2  = (const float*)d_in[2];
    const float* lam = (const float*)d_in[3];
    float* out = (float*)d_out;

    // workspace layout (needs ~112 MB)
    constexpr size_t SZ_BT1 = (size_t)Nc * K1 * sizeof(__bf16);  // 16 MiB
    constexpr size_t SZ_BT2 = (size_t)Nc * K2 * sizeof(__bf16);  // 32 MiB
    constexpr size_t SZ_W1Y = (size_t)Bm * Nc * sizeof(float);   // 32 MiB
    constexpr size_t SZ_XBF = (size_t)Bm * K2 * sizeof(__bf16);  // 16 MiB
    char* ws = (char*)d_ws;
    __bf16* BT1 = (__bf16*)(ws);
    __bf16* BT2 = (__bf16*)(ws + SZ_BT1);
    float*  W1y = (float*)(ws + SZ_BT1 + SZ_BT2);
    __bf16* xA  = (__bf16*)(ws + SZ_BT1 + SZ_BT2 + SZ_W1Y);
    __bf16* xB  = (__bf16*)(ws + SZ_BT1 + SZ_BT2 + SZ_W1Y + SZ_XBF);

    // 1) sign-permuted bf16 weights (recomputed every call: stateless)
    {
        int tot1 = Nc * K1;
        prep_weights_kernel<<<tot1 / 256, 256, 0, stream>>>(W1, BT1, IN, K1, tot1);
        int tot2 = Nc * K2;
        prep_weights_kernel<<<tot2 / 256, 256, 0, stream>>>(W2, BT2, HID, K2, tot2);
    }

    // 2) y -> bf16 (natural [B, n*8+i] layout is already the GEMM A layout)
    {
        int tot = Bm * K1;
        f32_to_bf16_kernel<<<tot / 256, 256, 0, stream>>>(y, xA, tot);
    }

    // 3) W1y = y * W1big    (f32 result kept for all layers)
    dim3 grid(Nc / TN, Bm / TM);
    gemm_wmma_bf16<<<grid, 256, 0, stream>>>(xA, BT1, nullptr, W1y, nullptr,
                                             nullptr, Bm, Nc, K1, 0);

    // 4) layer 0: x = softthr(W1y)
    {
        int tot = Bm * Nc;
        lista_first_kernel<<<tot / 256, 256, 0, stream>>>(W1y, lam, out, xA, tot);
    }

    // 5) layers 1..4: x = softthr(W1y + x * W2big), ping-pong bf16 activations
    __bf16* cur = xA;
    __bf16* nxt = xB;
    for (int l = 1; l < NLAYERS; ++l) {
        gemm_wmma_bf16<<<grid, 256, 0, stream>>>(cur, BT2, W1y, out, nxt,
                                                 lam, Bm, Nc, K2, 1);
        __bf16* t = cur; cur = nxt; nxt = t;
    }
    (void)in_sizes; (void)n_in; (void)out_size; (void)ws_size;
}
